// ParallelGraphNet_21930103014269
// MI455X (gfx1250) — matrix operand order repaired and checked by reading, not compiled, not run on hardware
//
#include <hip/hip_runtime.h>
#include <math.h>

// CDNA5 / gfx1250: wave32, WMMA f32_16x16x32_f16.
typedef __attribute__((ext_vector_type(16))) _Float16 v16h;
typedef __attribute__((ext_vector_type(2)))  __fp16   v2fp16;   // return type of cvt_pkrtz
typedef __attribute__((ext_vector_type(8)))  float    v8f;

union V16H { v16h v; uint4 q[2]; };

// single v_cvt_pk_rtz_f16_f32
__device__ __forceinline__ unsigned pk2(float a, float b) {
  union { v2fp16 h; unsigned u; } t;
  t.h = __builtin_amdgcn_cvt_pkrtz(a, b);
  return t.u;
}

// ---------------- small helper kernels ----------------

__global__ void init_kernel(float* deg, float* dv, int N, int NC) {
  int i = blockIdx.x * blockDim.x + threadIdx.x;
  if (i < NC) dv[i] = 0.f;
  if (i < N)  deg[i] = 1.f;          // self-loop contribution to degree
}

__global__ void degree_kernel(const int* __restrict__ J, float* deg, int E) {
  int e = blockIdx.x * blockDim.x + threadIdx.x;
  if (e < E) atomicAdd(&deg[J[e]], 1.f);
}

__global__ void dinv_kernel(float* deg, int N) {
  int i = blockIdx.x * blockDim.x + threadIdx.x;
  if (i < N) deg[i] = rsqrtf(deg[i]);   // deg >= 1 always
}

__global__ void w_kernel(const int* __restrict__ I, const int* __restrict__ J,
                         const float* __restrict__ dinv, float* w, int E) {
  int e = blockIdx.x * blockDim.x + threadIdx.x;
  if (e < E) w[e] = dinv[I[e]] * dinv[J[e]];
}

// x[n][o] = relu(sum_i K1[o][i] * xn[i][n]),  node-major output
__global__ void x0_kernel(const float* __restrict__ xn, const float* __restrict__ K1,
                          float* __restrict__ x, int N) {
  __shared__ float k1s[64 * 16];
  for (int t = threadIdx.x; t < 64 * 16; t += blockDim.x) k1s[t] = K1[t];
  __syncthreads();
  int nd = blockIdx.x * blockDim.x + threadIdx.x;
  if (nd >= N) return;
  float in[16];
#pragma unroll
  for (int i = 0; i < 16; ++i) in[i] = xn[(size_t)i * N + nd];
#pragma unroll 8
  for (int o = 0; o < 64; ++o) {
    float s = 0.f;
#pragma unroll
    for (int i = 0; i < 16; ++i) s = fmaf(k1s[o * 16 + i], in[i], s);
    x[(size_t)nd * 64 + o] = fmaxf(s, 0.f);
  }
}

// x -= dt*H*div ; div = 0   (dt*H = 0.25*0.1 = 0.025)
__global__ void update_kernel(float* __restrict__ x, float* __restrict__ dv, int NC) {
  int i = blockIdx.x * blockDim.x + threadIdx.x;
  if (i < NC) { x[i] = fmaf(-0.025f, dv[i], x[i]); dv[i] = 0.f; }
}

// out[n][o] = log_softmax_o( sum_i KNc[o][i] * x[n][i] )
__global__ void out_kernel(const float* __restrict__ x, const float* __restrict__ KNc,
                           float* __restrict__ out, int N) {
  __shared__ float ks[64 * 64];
  for (int t = threadIdx.x; t < 64 * 64; t += blockDim.x) ks[t] = KNc[t];
  __syncthreads();
  int nd = blockIdx.x * blockDim.x + threadIdx.x;
  if (nd >= N) return;
  float xv[64];
#pragma unroll
  for (int i = 0; i < 64; ++i) xv[i] = x[(size_t)nd * 64 + i];
  float mx = -INFINITY;
  for (int o = 0; o < 64; ++o) {
    float s = 0.f;
#pragma unroll
    for (int i = 0; i < 64; ++i) s = fmaf(ks[o * 64 + i], xv[i], s);
    mx = fmaxf(mx, s);
  }
  float se = 0.f;
  for (int o = 0; o < 64; ++o) {
    float s = 0.f;
#pragma unroll
    for (int i = 0; i < 64; ++i) s = fmaf(ks[o * 64 + i], xv[i], s);
    se += __expf(s - mx);
  }
  float lse = __logf(se);
  for (int o = 0; o < 64; ++o) {
    float s = 0.f;
#pragma unroll
    for (int i = 0; i < 64; ++i) s = fmaf(ks[o * 64 + i], xv[i], s);
    out[(size_t)nd * 64 + o] = s - mx - lse;
  }
}

// ---------------- fused WMMA edge kernel ----------------
// One wave = tile of 16 edges.
//   G (64ch x 16e)  -> Z1 = relu(KN1 * G) -> Z2 = KN1^T * Z1 -> atomic scatter +/- w*Z2
// B operands for matmul-1 are built DIRECTLY in registers: lane (n,h) gathers exactly
// channels [kc*32+16h, +16) of edge n, i.e. the contiguous K-stripe (k = 16h+s) that A's order matches.
// Only Z1 needs an LDS round-trip (D-layout half-M <-> B-layout half-K crosses lanes).
__global__ __launch_bounds__(256)
void edge_kernel(const int* __restrict__ I, const int* __restrict__ J,
                 const float* __restrict__ w, const float* __restrict__ KN1,
                 const float* __restrict__ x, float* __restrict__ dv, int E) {
  __shared__ alignas(16) _Float16 z_lds[8][16 * 64];  // per-wave [edge][ch] staging
  const int lane = threadIdx.x & 31;
  const int wid  = threadIdx.x >> 5;
  const int n = lane & 15;   // edge within tile (B/D column; A row index)
  const int h = lane >> 4;   // lane-half (K-half for A/B, M-half for D)
  _Float16* zsl = z_lds[wid];

  // Preload A operands (contiguous k order, matching B; not the ISA table order):
  // VGPR v, half h, sub s:  k = 16h+2v+s
  v16h a1[8], a2[8];        // a1: KN1 blocks, a2: KN1^T blocks; [m*2+kc]
#pragma unroll
  for (int m = 0; m < 4; ++m) {
#pragma unroll
    for (int kc = 0; kc < 2; ++kc) {
#pragma unroll
      for (int v = 0; v < 8; ++v) {
#pragma unroll
        for (int s = 0; s < 2; ++s) {
          int k   = 16 * h + 2 * v + s;
          int col = kc * 32 + k;
          int row = m * 16 + n;
          a1[m * 2 + kc][2 * v + s] = (_Float16)KN1[row * 64 + col];  // KN1[row][col]
          a2[m * 2 + kc][2 * v + s] = (_Float16)KN1[col * 64 + row];  // KN1^T[row][col]
        }
      }
    }
  }

  const int nTiles  = (E + 15) >> 4;
  const int wgWaves = blockDim.x >> 5;
  const int gwid    = blockIdx.x * wgWaves + wid;
  const int stride  = gridDim.x * wgWaves;

  for (int tile = gwid; tile < nTiles; tile += stride) {
    int  e     = tile * 16 + n;
    bool valid = (e < E);
    int   ii = valid ? I[e] : 0;
    int   jj = valid ? J[e] : 0;
    float we = valid ? w[e] : 0.f;          // tail lanes contribute 0 everywhere

    // Gather + build B operands in registers.
    // b[kc] halfs (in order) = we*(xI - xJ) for channels kc*32 + 16h + [0,16).
    V16H b[2];
#pragma unroll
    for (int kc = 0; kc < 2; ++kc) {
      const float4* pI = (const float4*)(x + (size_t)ii * 64 + kc * 32 + 16 * h);
      const float4* pJ = (const float4*)(x + (size_t)jj * 64 + kc * 32 + 16 * h);
      float4 i0 = pI[0], i1 = pI[1], i2 = pI[2], i3 = pI[3];
      float4 j0 = pJ[0], j1 = pJ[1], j2 = pJ[2], j3 = pJ[3];
      uint4 q0, q1;
      q0.x = pk2(we * (i0.x - j0.x), we * (i0.y - j0.y));
      q0.y = pk2(we * (i0.z - j0.z), we * (i0.w - j0.w));
      q0.z = pk2(we * (i1.x - j1.x), we * (i1.y - j1.y));
      q0.w = pk2(we * (i1.z - j1.z), we * (i1.w - j1.w));
      q1.x = pk2(we * (i2.x - j2.x), we * (i2.y - j2.y));
      q1.y = pk2(we * (i2.z - j2.z), we * (i2.w - j2.w));
      q1.z = pk2(we * (i3.x - j3.x), we * (i3.y - j3.y));
      q1.w = pk2(we * (i3.z - j3.z), we * (i3.w - j3.w));
      b[kc].q[0] = q0; b[kc].q[1] = q1;
    }

    v8f acc[4];
#pragma unroll
    for (int m = 0; m < 4; ++m) {
      v8f c = {0.f, 0.f, 0.f, 0.f, 0.f, 0.f, 0.f, 0.f};
      c = __builtin_amdgcn_wmma_f32_16x16x32_f16(false, a1[2 * m],     false, b[0].v, (short)0, c, false, false);
      c = __builtin_amdgcn_wmma_f32_16x16x32_f16(false, a1[2 * m + 1], false, b[1].v, (short)0, c, false, false);
      acc[m] = c;
    }

    // relu + restage Z1 (D layout: VGPR v -> M = v + 8h within block m)
#pragma unroll
    for (int m = 0; m < 4; ++m) {
      uint4 q;
      q.x = pk2(fmaxf(acc[m][0], 0.f), fmaxf(acc[m][1], 0.f));
      q.y = pk2(fmaxf(acc[m][2], 0.f), fmaxf(acc[m][3], 0.f));
      q.z = pk2(fmaxf(acc[m][4], 0.f), fmaxf(acc[m][5], 0.f));
      q.w = pk2(fmaxf(acc[m][6], 0.f), fmaxf(acc[m][7], 0.f));
      *(uint4*)&zsl[n * 64 + 16 * m + 8 * h] = q;
    }
    __builtin_amdgcn_wave_barrier();        // wave-internal LDS handoff (DS is in-order)

    V16H c0, c1;
    const uint4* t0 = (const uint4*)&zsl[n * 64 + 16 * h];
    const uint4* t1 = (const uint4*)&zsl[n * 64 + 32 + 16 * h];
    c0.q[0] = t0[0]; c0.q[1] = t0[1];
    c1.q[0] = t1[0]; c1.q[1] = t1[1];

    v8f acc2[4];
#pragma unroll
    for (int m = 0; m < 4; ++m) {
      v8f c = {0.f, 0.f, 0.f, 0.f, 0.f, 0.f, 0.f, 0.f};
      c = __builtin_amdgcn_wmma_f32_16x16x32_f16(false, a2[2 * m],     false, c0.v, (short)0, c, false, false);
      c = __builtin_amdgcn_wmma_f32_16x16x32_f16(false, a2[2 * m + 1], false, c1.v, (short)0, c, false, false);
      acc2[m] = c;
    }

    // scatter div[I] += w*z2, div[J] -= w*z2  (L2-resident f32 atomics)
    float* dI = dv + (size_t)ii * 64;
    float* dJ = dv + (size_t)jj * 64;
#pragma unroll
    for (int m = 0; m < 4; ++m) {
#pragma unroll
      for (int v = 0; v < 8; ++v) {
        int   ch  = 16 * m + 8 * h + v;
        float val = we * acc2[m][v];
        atomicAdd(&dI[ch],  val);
        atomicAdd(&dJ[ch], -val);
      }
    }
    __builtin_amdgcn_wave_barrier();        // keep next tile's z-store after this tile's z-loads
  }
}

// ---------------- host launcher ----------------

extern "C" void kernel_launch(void* const* d_in, const int* in_sizes, int n_in,
                              void* d_out, int out_size, void* d_ws, size_t ws_size,
                              hipStream_t stream) {
  const float* xn  = (const float*)d_in[0];   // (1,16,N)
  const int*   I   = (const int*)  d_in[1];   // (E,)
  const int*   J   = (const int*)  d_in[2];   // (E,)
  const float* K1  = (const float*)d_in[4];   // (64,16)
  const float* KN1 = (const float*)d_in[5];   // (64,64)
  const float* KNc = (const float*)d_in[6];   // (64,64)
  float* out = (float*)d_out;                 // (N,64)

  const int N  = in_sizes[0] / 16;
  const int E  = in_sizes[1];
  const int NC = N * 64;

  // workspace layout (floats, 16B aligned chunks)
  float* deg = (float*)d_ws;                   // N   (becomes dinv in place)
  float* w   = deg + ((N + 3) & ~3);           // E
  float* x   = w   + ((E + 3) & ~3);           // 64N node-major features
  float* dv  = x   + (size_t)NC;               // 64N divergence accumulator

  const int T = 256;
  init_kernel  <<<(NC + T - 1) / T, T, 0, stream>>>(deg, dv, N, NC);
  degree_kernel<<<(E  + T - 1) / T, T, 0, stream>>>(J, deg, E);
  dinv_kernel  <<<(N  + T - 1) / T, T, 0, stream>>>(deg, N);
  w_kernel     <<<(E  + T - 1) / T, T, 0, stream>>>(I, J, deg, w, E);
  x0_kernel    <<<(N  + T - 1) / T, T, 0, stream>>>(xn, K1, x, N);

  int nTiles = (E + 15) >> 4;
  int blocks = (nTiles + 7) / 8;
  if (blocks > 4096) blocks = 4096;
  if (blocks < 1) blocks = 1;

  for (int l = 0; l < 4; ++l) {
    edge_kernel  <<<blocks, 256, 0, stream>>>(I, J, w, KN1, x, dv, E);
    update_kernel<<<(NC + T - 1) / T, T, 0, stream>>>(x, dv, NC);
  }
  out_kernel<<<(N + T - 1) / T, T, 0, stream>>>(x, KNc, out, N);
}
